// DGCNN_gpvn_74071005987489
// MI455X (gfx1250) — compile-verified, matrix operand order checked by reading, not run on hardware
//
#include <hip/hip_runtime.h>
#include <hip/hip_bf16.h>
#include <math.h>

#define KNN 20
#define GROUPS 32
#define GN_EPS 1e-5f
#define LSLOPE 0.2f

typedef _Float16 h16 __attribute__((ext_vector_type(16)));
typedef _Float16 h8  __attribute__((ext_vector_type(8)));
typedef float    f8  __attribute__((ext_vector_type(8)));
typedef float    f4  __attribute__((ext_vector_type(4)));
typedef int      i4v __attribute__((ext_vector_type(4)));
typedef int      i8v __attribute__((ext_vector_type(8)));

// ---------------------------------------------------------------------------
// GEMM: Y[O,S] = W[O,ldW](cols 0..C) x X[C,S], f16 WMMA, f32 accumulate.
// Block tile 64(O) x 128(S), K-step 32. 8 wave32 waves as 4(O) x 2(S);
// each wave: one A fragment reused across 4 B fragments -> 4 WMMA / K-step.
// Every O is a multiple of 64 and every S a multiple of 128 in this network,
// so only the K tail (C % 32: the C=12 and C=3 cases) needs guards.
// LDS tiles are K-major so each WMMA fragment is exactly 2x ds_load_b128.
// ---------------------------------------------------------------------------
#define BO 64
#define BS 128
#define BK 32
#define KPAD 40   // row stride in halves (80 B, 16B aligned, conflict padding)

__global__ __launch_bounds__(256) void gemm_wmma_f16(
    const float* __restrict__ W, int ldW,
    const float* __restrict__ X,
    float* __restrict__ Y, int O, int C, int S)
{
    __shared__ __align__(16) _Float16 Wl[BO][KPAD];   // [o][c]
    __shared__ __align__(16) _Float16 Xl[BS][KPAD];   // [s][c]  (transposed)
    const int o0 = blockIdx.y * BO;
    const int s0 = blockIdx.x * BS;
    const int tid  = threadIdx.x;
    const int lane = tid & 31;
    const int wv   = tid >> 5;
    const int wo   = wv & 3;                 // 4 O-tiles of 16
    const int ws   = wv >> 2;                // 2 S-macrotiles of 64
    const int col  = lane & 15;
    const int khalf = (lane >> 4) << 3;      // 16-bit A/B striping: K offset 0|8
    const int mofs  = (lane >> 4) << 3;      // C/D: lanes 16-31 hold M=8..15

    // W loader: o = tid>>2 (0..63), c = (tid&3)*8
    const int lwo = tid >> 2, lwc = (tid & 3) << 3;
    // X loader: s = tid&127, c-half = (tid>>7)*16
    const int lxs = tid & 127, lxc = (tid >> 7) << 4;

    f8 acc0 = {0.f,0.f,0.f,0.f,0.f,0.f,0.f,0.f};
    f8 acc1 = acc0, acc2 = acc0, acc3 = acc0;

    const int arow = (wo << 4) + col;
    auto mma_all = [&]() {
        i4v w0 = *(const i4v*)&Wl[arow][khalf];
        i4v w1 = *(const i4v*)&Wl[arow][16 + khalf];
        i8v ai;
        ai[0]=w0[0]; ai[1]=w0[1]; ai[2]=w0[2]; ai[3]=w0[3];
        ai[4]=w1[0]; ai[5]=w1[1]; ai[6]=w1[2]; ai[7]=w1[3];
        h16 a = __builtin_bit_cast(h16, ai);
        auto bstep = [&](int t, f8& accv) {
            const int srow = (ws << 6) + (t << 4) + col;
            i4v x0 = *(const i4v*)&Xl[srow][khalf];
            i4v x1 = *(const i4v*)&Xl[srow][16 + khalf];
            i8v bi;
            bi[0]=x0[0]; bi[1]=x0[1]; bi[2]=x0[2]; bi[3]=x0[3];
            bi[4]=x1[0]; bi[5]=x1[1]; bi[6]=x1[2]; bi[7]=x1[3];
            h16 b = __builtin_bit_cast(h16, bi);
            accv = __builtin_amdgcn_wmma_f32_16x16x32_f16(
                false, a, false, b, (short)0, accv, false, false);
        };
        bstep(0, acc0); bstep(1, acc1); bstep(2, acc2); bstep(3, acc3);
    };

    const int kfull = C & ~(BK - 1);
    for (int c0 = 0; c0 < kfull; c0 += BK) {
        // ---- guard-free vectorized staging ----
        {
            const float* wp = &W[(size_t)(o0 + lwo) * ldW + c0 + lwc];
            f4 a0 = *(const f4*)(wp);
            f4 a1 = *(const f4*)(wp + 4);
            h8 hw;
            hw[0]=(_Float16)a0[0]; hw[1]=(_Float16)a0[1];
            hw[2]=(_Float16)a0[2]; hw[3]=(_Float16)a0[3];
            hw[4]=(_Float16)a1[0]; hw[5]=(_Float16)a1[1];
            hw[6]=(_Float16)a1[2]; hw[7]=(_Float16)a1[3];
            *(h8*)&Wl[lwo][lwc] = hw;

            float v[16];
            const float* xp = &X[(size_t)(c0 + lxc) * S + s0 + lxs];
            #pragma unroll
            for (int i = 0; i < 16; ++i) v[i] = xp[(size_t)i * S];
            h8 h0, h1;
            #pragma unroll
            for (int i = 0; i < 8; ++i) h0[i] = (_Float16)v[i];
            #pragma unroll
            for (int i = 0; i < 8; ++i) h1[i] = (_Float16)v[8 + i];
            *(h8*)&Xl[lxs][lxc]     = h0;
            *(h8*)&Xl[lxs][lxc + 8] = h1;
        }
        if (c0 + BK < kfull) {
            // pull next X tile toward caches (global_prefetch_b8 on gfx1250)
            __builtin_prefetch(&X[(size_t)(c0 + BK + lxc) * S + s0 + lxs], 0, 1);
        }
        __syncthreads();
        mma_all();
        __syncthreads();
    }

    if (kfull < C) {  // K tail (C % 32 != 0: the C=12 / C=3 layers)
        for (int i = tid; i < BO * BK; i += 256) {
            int o = i >> 5, c = i & 31;
            float v = 0.f;
            if (kfull + c < C) v = W[(size_t)(o0 + o) * ldW + kfull + c];
            Wl[o][c] = (_Float16)v;
        }
        for (int i = tid; i < BS * BK; i += 256) {
            int s = i >> 5, c = i & 31;
            float v = 0.f;
            if (kfull + c < C) v = X[(size_t)(kfull + c) * S + s0 + s];
            Xl[s][c] = (_Float16)v;
        }
        __syncthreads();
        mma_all();
        __syncthreads();
    }

    // store (exact fit: no guards needed for this network's shapes)
    #pragma unroll
    for (int r = 0; r < 8; ++r) {
        int oo = o0 + (wo << 4) + mofs + r;
        size_t base = (size_t)oo * S + s0 + (ws << 6) + col;
        Y[base]      = acc0[r];
        Y[base + 16] = acc1[r];
        Y[base + 32] = acc2[r];
        Y[base + 48] = acc3[r];
    }
}

// ---------------------------------------------------------------------------
// kNN support: Gram matrix G = F^T F comes from the WMMA GEMM above.
// ---------------------------------------------------------------------------
// out[q*C + c] = F[c, qmap(q)]  (rows of F^T, optionally gathered)
__global__ void feat_transpose(const float* __restrict__ F, int C, int M, int Q,
                               const int* __restrict__ qmap, float* __restrict__ out)
{
    size_t t = (size_t)blockIdx.x * blockDim.x + threadIdx.x;
    if (t >= (size_t)Q * C) return;
    int q = (int)(t / C), c = (int)(t % C);
    int qp = qmap ? qmap[q] : q;
    out[t] = F[(size_t)c * M + qp];
}

// xx[m] = sum_c F[c,m]^2
__global__ void col_sumsq(const float* __restrict__ F, int C, int M, float* __restrict__ xx)
{
    int m = blockIdx.x * blockDim.x + threadIdx.x;
    if (m >= M) return;
    float s = 0.f;
    for (int c = 0; c < C; ++c) { float v = F[(size_t)c * M + m]; s = fmaf(v, v, s); }
    xx[m] = s;
}

// Row-parallel top-20 of pd[q,m] = 2*G[q,m] - xx[qmap(q)] - xx[m].
// One 256-thread block per query; per-thread sorted top-20 over a strided
// slice (stable ascending-m), then 20 rounds of block argmax merge with the
// jax.lax.top_k tie-break (value desc, index asc).
__global__ __launch_bounds__(256) void topk_rows(const float* __restrict__ G,
                                                 const float* __restrict__ xx, int M,
                                                 const int* __restrict__ qmap,
                                                 int* __restrict__ idx)
{
    int q = blockIdx.x;
    int qp = qmap ? qmap[q] : q;
    const float* row = G + (size_t)q * M;
    const float xq = xx[qp];
    float bv[KNN]; int bi[KNN];
    #pragma unroll
    for (int i = 0; i < KNN; ++i) { bv[i] = -3.402823e38f; bi[i] = 0x7fffffff; }
    for (int m = threadIdx.x; m < M; m += 256) {
        float d = 2.f * row[m] - xq - xx[m];
        if (d > bv[KNN - 1]) {
            int p = KNN - 1;
            while (p > 0 && bv[p - 1] < d) { bv[p] = bv[p - 1]; bi[p] = bi[p - 1]; --p; }
            bv[p] = d; bi[p] = m;
        }
    }
    __shared__ float sv[256];
    __shared__ int   si[256];
    int head = 0;
    for (int k = 0; k < KNN; ++k) {
        sv[threadIdx.x] = (head < KNN) ? bv[head] : -3.402823e38f;
        si[threadIdx.x] = (head < KNN) ? bi[head] : 0x7fffffff;
        __syncthreads();
        for (int s = 128; s > 0; s >>= 1) {
            if (threadIdx.x < s) {
                float v1 = sv[threadIdx.x], v2 = sv[threadIdx.x + s];
                int   i1 = si[threadIdx.x], i2 = si[threadIdx.x + s];
                bool take2 = (v2 > v1) || (v2 == v1 && i2 < i1);
                if (take2) { sv[threadIdx.x] = v2; si[threadIdx.x] = i2; }
            }
            __syncthreads();
        }
        float wvv = sv[0]; int wii = si[0];
        if (head < KNN && bv[head] == wvv && bi[head] == wii) ++head;  // winner advances
        __syncthreads();
        if (threadIdx.x == 0) idx[(size_t)q * KNN + k] = wii;
    }
}

// graph feature: out[c2, q*K+k] = c2<C ? f[c2,idx]-f[c2,qp] : f[c2-C,qp]
__global__ void graph_feat(const float* __restrict__ feat, int C, int M, int Q,
                           const int* __restrict__ qmap, const int* __restrict__ idx,
                           float* __restrict__ out)
{
    size_t t = (size_t)blockIdx.x * blockDim.x + threadIdx.x;
    size_t total = (size_t)2 * C * Q * KNN;
    if (t >= total) return;
    int qk = (int)(t % ((size_t)Q * KNN));
    int c2 = (int)(t / ((size_t)Q * KNN));
    int q = qk / KNN, k = qk % KNN;
    int qp = qmap ? qmap[q] : q;
    float v;
    if (c2 < C) v = feat[(size_t)c2 * M + idx[q * KNN + k]] - feat[(size_t)c2 * M + qp];
    else        v = feat[(size_t)(c2 - C) * M + qp];
    out[t] = v;
}

// GroupNorm statistics: one block per group (group channels are contiguous rows)
__global__ __launch_bounds__(256) void gn_stats(const float* __restrict__ Y, int O, int S,
                                                float* __restrict__ mean, float* __restrict__ rstd)
{
    int g = blockIdx.x;
    int cg = O / GROUPS;
    size_t n = (size_t)cg * S;
    const float* base = Y + (size_t)g * cg * S;
    float s1 = 0.f, s2 = 0.f;
    for (size_t i = threadIdx.x; i < n; i += 256) { float v = base[i]; s1 += v; s2 += v * v; }
    __shared__ float r1[256], r2[256];
    r1[threadIdx.x] = s1; r2[threadIdx.x] = s2;
    __syncthreads();
    for (int s = 128; s > 0; s >>= 1) {
        if (threadIdx.x < s) { r1[threadIdx.x] += r1[threadIdx.x + s]; r2[threadIdx.x] += r2[threadIdx.x + s]; }
        __syncthreads();
    }
    if (threadIdx.x == 0) {
        float m = r1[0] / (float)n;
        float var = r2[0] / (float)n - m * m;
        mean[g] = m;
        rstd[g] = rsqrtf(fmaxf(var, 0.f) + GN_EPS);
    }
}

__global__ void gn_apply(float* __restrict__ Y, int O, int S,
                         const float* __restrict__ gamma, const float* __restrict__ beta,
                         const float* __restrict__ mean, const float* __restrict__ rstd)
{
    size_t t = (size_t)blockIdx.x * blockDim.x + threadIdx.x;
    size_t tot = (size_t)O * S;
    if (t >= tot) return;
    int o = (int)(t / S);
    int g = o / (O / GROUPS);
    float v = (Y[t] - mean[g]) * rstd[g] * gamma[o] + beta[o];
    Y[t] = v >= 0.f ? v : LSLOPE * v;
}

__global__ void max_over_k(const float* __restrict__ in, int O, int Q, float* __restrict__ out)
{
    size_t t = (size_t)blockIdx.x * blockDim.x + threadIdx.x;
    size_t tot = (size_t)O * Q;
    if (t >= tot) return;
    const float* p = in + t * KNN;
    float m = p[0];
    #pragma unroll
    for (int k = 1; k < KNN; ++k) m = fmaxf(m, p[k]);
    out[t] = m;
}

__global__ void max_over_s(const float* __restrict__ in, int O, int S, float* __restrict__ out)
{
    int o = blockIdx.x * blockDim.x + threadIdx.x;
    if (o >= O) return;
    const float* p = in + (size_t)o * S;
    float m = p[0];
    for (int s = 1; s < S; ++s) m = fmaxf(m, p[s]);
    out[o] = m;
}

__global__ void bcast_rows(const float* __restrict__ vec, float* __restrict__ dst, int O, int S)
{
    size_t t = (size_t)blockIdx.x * blockDim.x + threadIdx.x;
    if (t >= (size_t)O * S) return;
    dst[t] = vec[t / S];
}

__global__ void gather_cols(const float* __restrict__ src, const int* __restrict__ fpi,
                            float* __restrict__ dst, int C, int M, int Q)
{
    size_t t = (size_t)blockIdx.x * blockDim.x + threadIdx.x;
    if (t >= (size_t)C * Q) return;
    int c = (int)(t / Q), j = (int)(t % Q);
    dst[t] = src[(size_t)c * M + fpi[j]];
}

// per-row sum / sumsq / max / min (one block per row)
__global__ __launch_bounds__(256) void row_stats(const float* __restrict__ A, int S,
                                                 float* __restrict__ s1, float* __restrict__ s2,
                                                 float* __restrict__ mx, float* __restrict__ mn)
{
    int o = blockIdx.x;
    const float* p = A + (size_t)o * S;
    float a = 0.f, b = 0.f, hi = -3.402823e38f, lo = 3.402823e38f;
    for (int i = threadIdx.x; i < S; i += 256) {
        float v = p[i]; a += v; b += v * v; hi = fmaxf(hi, v); lo = fminf(lo, v);
    }
    __shared__ float ra[256], rb[256], rh[256], rl[256];
    ra[threadIdx.x] = a; rb[threadIdx.x] = b; rh[threadIdx.x] = hi; rl[threadIdx.x] = lo;
    __syncthreads();
    for (int s = 128; s > 0; s >>= 1) {
        if (threadIdx.x < s) {
            ra[threadIdx.x] += ra[threadIdx.x + s];
            rb[threadIdx.x] += rb[threadIdx.x + s];
            rh[threadIdx.x] = fmaxf(rh[threadIdx.x], rh[threadIdx.x + s]);
            rl[threadIdx.x] = fminf(rl[threadIdx.x], rl[threadIdx.x + s]);
        }
        __syncthreads();
    }
    if (threadIdx.x == 0) { s1[o] = ra[0]; s2[o] = rb[0]; mx[o] = rh[0]; mn[o] = rl[0]; }
}

// GN stats for factorized layer Y[o,n,j] = A[o,n] + B[o,j] (closed form)
__global__ void fact_stats(const float* __restrict__ sA, const float* __restrict__ sA2,
                           const float* __restrict__ sB, const float* __restrict__ sB2,
                           int O, float Sn, float Sj,
                           float* __restrict__ mean, float* __restrict__ rstd)
{
    int g = threadIdx.x;
    if (g >= GROUPS) return;
    int cg = O / GROUPS;
    float s1 = 0.f, s2 = 0.f;
    for (int i = 0; i < cg; ++i) {
        int o = g * cg + i;
        s1 += Sj * sA[o] + Sn * sB[o];
        s2 += Sj * sA2[o] + 2.f * sA[o] * sB[o] + Sn * sB2[o];
    }
    float cnt = cg * Sn * Sj;
    float m = s1 / cnt;
    float var = s2 / cnt - m * m;
    mean[g] = m;
    rstd[g] = rsqrtf(fmaxf(var, 0.f) + GN_EPS);
}

// max_j lrelu(gn(A[o,n]+B[o,j])) == lrelu(gn(A[o,n] + Bext[o])), Bext by sign(gamma)
__global__ void hm_from_fact(const float* __restrict__ A,
                             const float* __restrict__ Bmax, const float* __restrict__ Bmin,
                             const float* __restrict__ gamma, const float* __restrict__ beta,
                             const float* __restrict__ mean, const float* __restrict__ rstd,
                             int O, int S, float* __restrict__ out)
{
    size_t t = (size_t)blockIdx.x * blockDim.x + threadIdx.x;
    if (t >= (size_t)O * S) return;
    int o = (int)(t / S);
    int g = o / (O / GROUPS);
    float Bx = gamma[o] >= 0.f ? Bmax[o] : Bmin[o];
    float v = (A[t] + Bx - mean[g]) * rstd[g] * gamma[o] + beta[o];
    out[t] = v >= 0.f ? v : LSLOPE * v;
}

// x_seg[q,n] = sum_o w9[q,o] * lrelu(gn(C8[o,n] + Dext[o]))   (O = 256 fixed)
__global__ void seg_out(const float* __restrict__ C8,
                        const float* __restrict__ Dmax, const float* __restrict__ Dmin,
                        const float* __restrict__ gamma, const float* __restrict__ beta,
                        const float* __restrict__ mean, const float* __restrict__ rstd,
                        const float* __restrict__ w9, float* __restrict__ out, int Nn)
{
    int n = blockIdx.x * blockDim.x + threadIdx.x;
    if (n >= Nn) return;
    float a0 = 0.f, a1 = 0.f;
    for (int o = 0; o < 256; ++o) {
        int g = o >> 3;
        float Dx = gamma[o] >= 0.f ? Dmax[o] : Dmin[o];
        float v = (C8[(size_t)o * Nn + n] + Dx - mean[g]) * rstd[g] * gamma[o] + beta[o];
        v = v >= 0.f ? v : LSLOPE * v;
        a0 = fmaf(w9[o], v, a0);
        a1 = fmaf(w9[256 + o], v, a1);
    }
    out[n] = a0;
    out[Nn + n] = a1;
}

// x_key[j,n] = sum_o w10[o] * lrelu(gn(C8[o,n] + D8[o,j]))    (O = 256 fixed)
__global__ void key_out(const float* __restrict__ C8, const float* __restrict__ D8,
                        const float* __restrict__ gamma, const float* __restrict__ beta,
                        const float* __restrict__ mean, const float* __restrict__ rstd,
                        const float* __restrict__ w10, float* __restrict__ out, int Nn, int NKk)
{
    int n = blockIdx.x * blockDim.x + threadIdx.x;
    int j = blockIdx.y;
    if (n >= Nn) return;
    float acc = 0.f;
    for (int o = 0; o < 256; ++o) {
        int g = o >> 3;
        float v = (C8[(size_t)o * Nn + n] + D8[(size_t)o * NKk + j] - mean[g]) * rstd[g] * gamma[o] + beta[o];
        v = v >= 0.f ? v : LSLOPE * v;
        acc = fmaf(w10[o], v, acc);
    }
    out[(size_t)j * Nn + n] = acc;
}

// ---------------------------------------------------------------------------
extern "C" void kernel_launch(void* const* d_in, const int* in_sizes, int n_in,
                              void* d_out, int out_size, void* d_ws, size_t ws_size,
                              hipStream_t stream)
{
    (void)in_sizes; (void)n_in; (void)out_size; (void)ws_size;
    const int N = 2048, NKEY = 128;

    const float* x     = (const float*)d_in[0];
    const float* objpc = (const float*)d_in[1];
    const int*   fpi   = (const int*)d_in[2];
    const float* WO1 = (const float*)d_in[3];  const float* WO2 = (const float*)d_in[4];
    const float* WO3 = (const float*)d_in[5];  const float* WO4 = (const float*)d_in[6];
    const float* WO5 = (const float*)d_in[7];  const float* WO6 = (const float*)d_in[8];
    const float* WO7 = (const float*)d_in[9];  const float* WO8 = (const float*)d_in[10];
    const float* W1  = (const float*)d_in[11]; const float* W2  = (const float*)d_in[12];
    const float* W3  = (const float*)d_in[13]; const float* W4  = (const float*)d_in[14];
    const float* W5  = (const float*)d_in[15]; const float* W6  = (const float*)d_in[16];
    const float* W7  = (const float*)d_in[17]; const float* W8  = (const float*)d_in[18];
    const float* W8R = (const float*)d_in[19]; const float* W9  = (const float*)d_in[20];
    const float* W10 = (const float*)d_in[21];
    const float* GO1 = (const float*)d_in[22]; const float* BO1 = (const float*)d_in[23];
    const float* GO2 = (const float*)d_in[24]; const float* BO2 = (const float*)d_in[25];
    const float* GO3 = (const float*)d_in[26]; const float* BO3 = (const float*)d_in[27];
    const float* GO4 = (const float*)d_in[28]; const float* BO4 = (const float*)d_in[29];
    const float* GO5 = (const float*)d_in[30]; const float* BO5 = (const float*)d_in[31];
    const float* GO6 = (const float*)d_in[32]; const float* BO6 = (const float*)d_in[33];
    const float* GO7 = (const float*)d_in[34]; const float* BO7 = (const float*)d_in[35];
    const float* GO8 = (const float*)d_in[36]; const float* BO8 = (const float*)d_in[37];
    const float* G1  = (const float*)d_in[38]; const float* B1  = (const float*)d_in[39];
    const float* G2  = (const float*)d_in[40]; const float* B2  = (const float*)d_in[41];
    const float* G3  = (const float*)d_in[42]; const float* B3  = (const float*)d_in[43];
    const float* G4  = (const float*)d_in[44]; const float* B4  = (const float*)d_in[45];
    const float* G5  = (const float*)d_in[46]; const float* B5  = (const float*)d_in[47];
    const float* G6  = (const float*)d_in[48]; const float* B6  = (const float*)d_in[49];
    const float* G7  = (const float*)d_in[50]; const float* B7  = (const float*)d_in[51];
    const float* G8  = (const float*)d_in[52]; const float* B8  = (const float*)d_in[53];
    const float* G8R = (const float*)d_in[54]; const float* B8R = (const float*)d_in[55];

    // ---- workspace bump allocator ----
    char* wsb = (char*)d_ws;
    size_t off = 0;
    auto alloc_f = [&](size_t n) -> float* {
        float* p = (float*)(wsb + off);
        off += ((n * sizeof(float) + 255) & ~(size_t)255);
        return p;
    };
    auto alloc_i = [&](size_t n) -> int* {
        int* p = (int*)(wsb + off);
        off += ((n * sizeof(int) + 255) & ~(size_t)255);
        return p;
    };

    float* F    = alloc_f((size_t)128 * N * KNN);   // graph features (max 128 x 40960)
    float* Y1   = alloc_f((size_t)64 * N * KNN);    // gemm ping
    float* Y2   = alloc_f((size_t)64 * N * KNN);    // gemm pong
    float* cat1216  = alloc_f((size_t)1216 * N);    // [hg bcast | x1 | x2 | x3]
    float* cat1216o = alloc_f((size_t)1216 * NKEY); // [og bcast | obj1 | obj2 | obj3]
    float* H6   = alloc_f((size_t)1024 * N);
    float* H7   = alloc_f((size_t)512 * N);
    float* obj1_full = alloc_f((size_t)64 * N);
    float* obj_small = alloc_f((size_t)128 * NKEY);
    float* og = alloc_f(1024);
    float* hg = alloc_f(1024);
    float* A8  = alloc_f((size_t)256 * N);
    float* Bm8 = alloc_f((size_t)256 * NKEY);
    float* HM  = alloc_f((size_t)256 * N);
    float* C8  = alloc_f((size_t)256 * N);
    float* D8  = alloc_f((size_t)256 * NKEY);
    float* Gm  = alloc_f((size_t)N * N);            // Gram matrix for kNN
    float* FT  = alloc_f((size_t)N * 64);           // F^T rows for kNN GEMM
    float* xxb = alloc_f(N);                        // per-point squared norms
    float* stg  = alloc_f(64);   // generic GN stats (mean|rstd)
    float* st8  = alloc_f(64);   // w8 layer stats
    float* st8r = alloc_f(64);   // w8r layer stats
    float* sA  = alloc_f(256); float* sA2 = alloc_f(256);
    float* sAx = alloc_f(256); float* sAn = alloc_f(256);
    float* sB  = alloc_f(256); float* sB2 = alloc_f(256);
    float* sBx = alloc_f(256); float* sBn = alloc_f(256);
    int* idxN = alloc_i((size_t)N * KNN);
    int* idxK = alloc_i((size_t)NKEY * KNN);

    float* x1 = cat1216 + (size_t)1024 * N;
    float* x2 = x1 + (size_t)64 * N;
    float* x3 = x2 + (size_t)64 * N;
    float* obj1 = cat1216o + (size_t)1024 * NKEY;
    float* obj2 = obj1 + (size_t)64 * NKEY;
    float* obj3 = obj2 + (size_t)64 * NKEY;

    // ---- launch helpers ----
    auto gemm = [&](const float* Wp, int ldW, const float* Xp, float* Yp, int O, int C, int S) {
        dim3 g((unsigned)((S + BS - 1) / BS), (unsigned)((O + BO - 1) / BO));
        gemm_wmma_f16<<<g, dim3(256), 0, stream>>>(Wp, ldW, Xp, Yp, O, C, S);
    };
    auto cblock = [&](const float* Wp, int ldW, const float* Xp, float* Yp, int O, int C, int S,
                      const float* gam, const float* bet) {
        gemm(Wp, ldW, Xp, Yp, O, C, S);
        gn_stats<<<dim3(GROUPS), dim3(256), 0, stream>>>(Yp, O, S, stg, stg + GROUPS);
        size_t tot = (size_t)O * S;
        gn_apply<<<dim3((unsigned)((tot + 255) / 256)), dim3(256), 0, stream>>>(
            Yp, O, S, gam, bet, stg, stg + GROUPS);
    };
    // kNN: G = F^T F via WMMA GEMM, then row-parallel top-20
    auto knn = [&](const float* f, int C, int M, int Q, const int* qm, int* idx) {
        size_t tt = (size_t)Q * C;
        feat_transpose<<<dim3((unsigned)((tt + 255) / 256)), dim3(256), 0, stream>>>(f, C, M, Q, qm, FT);
        gemm(FT, C, f, Gm, Q, C, M);
        col_sumsq<<<dim3((unsigned)((M + 255) / 256)), dim3(256), 0, stream>>>(f, C, M, xxb);
        topk_rows<<<dim3((unsigned)Q), dim3(256), 0, stream>>>(Gm, xxb, M, qm, idx);
    };
    auto gfeat = [&](const float* f, int C, int M, int Q, const int* qm, const int* idx, float* o) {
        size_t tot = (size_t)2 * C * Q * KNN;
        graph_feat<<<dim3((unsigned)((tot + 255) / 256)), dim3(256), 0, stream>>>(f, C, M, Q, qm, idx, o);
    };
    auto maxk = [&](const float* in, int O, int Q, float* o) {
        size_t tot = (size_t)O * Q;
        max_over_k<<<dim3((unsigned)((tot + 255) / 256)), dim3(256), 0, stream>>>(in, O, Q, o);
    };

    // ======================= object branch =======================
    knn(objpc, 3, N, N, nullptr, idxN);
    gfeat(objpc, 6, N, N, nullptr, idxN, F);
    cblock(WO1, 12, F, Y1, 64, 12, N * KNN, GO1, BO1);
    cblock(WO2, 64, Y1, Y2, 64, 64, N * KNN, GO2, BO2);
    maxk(Y2, 64, N, obj1_full);

    knn(obj1_full, 64, N, NKEY, fpi, idxK);                // only fpi rows are used downstream
    gfeat(obj1_full, 64, N, NKEY, fpi, idxK, F);
    gather_cols<<<dim3((64 * NKEY + 255) / 256), dim3(256), 0, stream>>>(obj1_full, fpi, obj1, 64, N, NKEY);
    cblock(WO3, 128, F, Y1, 64, 128, NKEY * KNN, GO3, BO3);
    cblock(WO4, 64, Y1, Y2, 64, 64, NKEY * KNN, GO4, BO4);
    maxk(Y2, 64, NKEY, obj2);

    knn(obj2, 64, NKEY, NKEY, nullptr, idxK);
    gfeat(obj2, 64, NKEY, NKEY, nullptr, idxK, F);
    cblock(WO5, 128, F, Y1, 64, 128, NKEY * KNN, GO5, BO5);
    maxk(Y1, 64, NKEY, obj3);

    cblock(WO6, 192, obj1 /*=cat192_o*/, Y1, 1024, 192, NKEY, GO6, BO6);
    max_over_s<<<dim3(4), dim3(256), 0, stream>>>(Y1, 1024, NKEY, og);
    bcast_rows<<<dim3((1024 * NKEY + 255) / 256), dim3(256), 0, stream>>>(og, cat1216o, 1024, NKEY);
    cblock(WO7, 1216, cat1216o, Y2, 512, 1216, NKEY, GO7, BO7);
    cblock(WO8, 512, Y2, obj_small, 128, 512, NKEY, GO8, BO8);

    // ======================= x branch =======================
    knn(x, 3, N, N, nullptr, idxN);
    gfeat(x, 6, N, N, nullptr, idxN, F);
    cblock(W1, 12, F, Y1, 64, 12, N * KNN, G1, B1);
    cblock(W2, 64, Y1, Y2, 64, 64, N * KNN, G2, B2);
    maxk(Y2, 64, N, x1);

    knn(x1, 64, N, N, nullptr, idxN);
    gfeat(x1, 64, N, N, nullptr, idxN, F);
    cblock(W3, 128, F, Y1, 64, 128, N * KNN, G3, B3);
    cblock(W4, 64, Y1, Y2, 64, 64, N * KNN, G4, B4);
    maxk(Y2, 64, N, x2);

    knn(x2, 64, N, N, nullptr, idxN);
    gfeat(x2, 64, N, N, nullptr, idxN, F);
    cblock(W5, 128, F, Y1, 64, 128, N * KNN, G5, B5);
    maxk(Y1, 64, N, x3);

    cblock(W6, 192, x1 /*=cat192*/, H6, 1024, 192, N, G6, B6);
    max_over_s<<<dim3(4), dim3(256), 0, stream>>>(H6, 1024, N, hg);
    bcast_rows<<<dim3((unsigned)(((size_t)1024 * N + 255) / 256)), dim3(256), 0, stream>>>(hg, cat1216, 1024, N);
    cblock(W7, 1216, cat1216, H7, 512, 1216, N, G7, B7);

    // ---- factorized w8: h8[o,n,j] = A8[o,n] + Bm8[o,j] (pre-GN) ----
    gemm(W8, 640, H7, A8, 256, 512, N);                   // W8[:, :512]  x H7
    gemm(W8 + 512, 640, obj_small, Bm8, 256, 128, NKEY);  // W8[:, 512:] x obj
    row_stats<<<dim3(256), dim3(256), 0, stream>>>(A8, N, sA, sA2, sAx, sAn);
    row_stats<<<dim3(256), dim3(256), 0, stream>>>(Bm8, NKEY, sB, sB2, sBx, sBn);
    fact_stats<<<dim3(1), dim3(32), 0, stream>>>(sA, sA2, sB, sB2, 256, (float)N, (float)NKEY,
                                                 st8, st8 + GROUPS);
    hm_from_fact<<<dim3((unsigned)(((size_t)256 * N + 255) / 256)), dim3(256), 0, stream>>>(
        A8, sBx, sBn, G8, B8, st8, st8 + GROUPS, 256, N, HM);

    // ---- factorized w8r: h8r[o,n,j] = C8[o,n] + D8[o,j] (pre-GN) ----
    gemm(W8R, 384, HM, C8, 256, 256, N);                  // W8R[:, :256] x hm
    gemm(W8R + 256, 384, obj_small, D8, 256, 128, NKEY);  // W8R[:, 256:] x obj
    row_stats<<<dim3(256), dim3(256), 0, stream>>>(C8, N, sA, sA2, sAx, sAn);
    row_stats<<<dim3(256), dim3(256), 0, stream>>>(D8, NKEY, sB, sB2, sBx, sBn);
    fact_stats<<<dim3(1), dim3(32), 0, stream>>>(sA, sA2, sB, sB2, 256, (float)N, (float)NKEY,
                                                 st8r, st8r + GROUPS);

    // ---- outputs ----
    float* out_seg = (float*)d_out;
    float* out_key = out_seg + (size_t)2 * N;
    seg_out<<<dim3((N + 255) / 256), dim3(256), 0, stream>>>(
        C8, sBx, sBn, G8R, B8R, st8r, st8r + GROUPS, W9, out_seg, N);
    key_out<<<dim3((N + 255) / 256, NKEY), dim3(256), 0, stream>>>(
        C8, D8, G8R, B8R, st8r, st8r + GROUPS, W10, out_key, N, NKEY);
}
